// _ProposalLayer_71459665871444
// MI455X (gfx1250) — matrix operand order repaired and checked by reading, not compiled, not run on hardware
//
#include <hip/hip_runtime.h>
#include <hip/hip_bf16.h>

// ---------------------------------------------------------------------------
// Types for CDNA5 WMMA
// ---------------------------------------------------------------------------
typedef __bf16 bf16_t;
typedef bf16_t v16bf __attribute__((ext_vector_type(16)));
typedef float  v8f   __attribute__((ext_vector_type(8)));
typedef int    v4i   __attribute__((vector_size(16)));

union FragAB { v16bf v; float4 f4[2]; };

__device__ __forceinline__ unsigned short f2bf(float f) {
    unsigned int u = __float_as_uint(f);
    unsigned int r = u + 0x7FFFu + ((u >> 16) & 1u);   // round-to-nearest-even
    return (unsigned short)(r >> 16);
}
__device__ __forceinline__ unsigned int pack2(unsigned short lo, unsigned short hi) {
    return (unsigned int)lo | ((unsigned int)hi << 16);
}

// ---------------------------------------------------------------------------
// gfx1250 async global->LDS path (guarded; falls back to sync copy)
// Probe result: builtin exists, param 0 is `v4i*` (16-byte int vector pointer).
// ---------------------------------------------------------------------------
#if defined(__has_builtin)
#if __has_builtin(__builtin_amdgcn_global_load_async_to_lds_b128) && \
    __has_builtin(__builtin_amdgcn_s_wait_asynccnt)
#define HAVE_ASYNC 1
#endif
#endif
#ifndef HAVE_ASYNC
#define HAVE_ASYNC 0
#endif

__device__ __forceinline__ void copy16(const void* g, void* l) {
#if HAVE_ASYNC
    __builtin_amdgcn_global_load_async_to_lds_b128((v4i*)g, (v4i*)l, 0, 0);
#else
    *(float4*)l = *(const float4*)g;
#endif
}
__device__ __forceinline__ void wait_async() {
#if HAVE_ASYNC
    __builtin_amdgcn_s_wait_asynccnt(0);
#endif
}

// ---------------------------------------------------------------------------
// Problem constants
// ---------------------------------------------------------------------------
#define NB     300
#define CIN    256
#define HIN    38
#define WIN    38
#define HX     39
#define WX     39
#define M1     (HX*WX)       // 1521
#define CMID   128
#define K1     (CIN*4)       // 1024
#define M2     (38*38)       // 1444
#define K2     (CMID*4)      // 512
#define NOUT2  24
#define SCN    5776

// ---------------------------------------------------------------------------
// Kernel 0a: W1 fp32 -> bf16 [oc][K1]
// ---------------------------------------------------------------------------
__global__ void k_pack_w1(const float* __restrict__ W1, unsigned short* __restrict__ wbf1) {
    int i = blockIdx.x * blockDim.x + threadIdx.x;
    if (i < CMID * K1) wbf1[i] = f2bf(W1[i]);
}

// ---------------------------------------------------------------------------
// Kernel 0b: Wc(8,128,2,2)+Wb(16,128,2,2) -> bf16 [32][tap*128+c], rows 24..31 = 0
// ---------------------------------------------------------------------------
__global__ void k_pack_w2(const float* __restrict__ Wc, const float* __restrict__ Wb,
                          unsigned short* __restrict__ wbf2) {
    int i = blockIdx.x * blockDim.x + threadIdx.x;
    if (i >= 32 * K2) return;
    int oc  = i >> 9;
    int k   = i & 511;
    int tap = k >> 7;
    int c   = k & 127;
    float v = 0.f;
    if (oc < 8)       v = Wc[(oc * 128 + c) * 4 + tap];
    else if (oc < 24) v = Wb[((oc - 8) * 128 + c) * 4 + tap];
    wbf2[i] = f2bf(v);
}

// ---------------------------------------------------------------------------
// Kernel 1: conv1 implicit GEMM, bf16 WMMA, double-buffered LDS.
//   grid (12, 300), block 256 (8 waves). M-block 128, N 128, K 1024 (32 x 32-chunks)
// ---------------------------------------------------------------------------
__global__ void __launch_bounds__(256)
k_conv1(const float* __restrict__ feat, const unsigned short* __restrict__ wbf1,
        const float* __restrict__ b1, unsigned short* __restrict__ xout) {
    __shared__ __align__(16) unsigned short Alds[2][128 * 32];
    __shared__ __align__(16) unsigned short Blds[2][128 * 32];
    __shared__ float bias_s[128];

    const int b      = blockIdx.y;
    const int mblock = blockIdx.x;
    const int t      = threadIdx.x;
    const int wave   = t >> 5;
    const int lane   = t & 31;
    const int lrow   = lane & 15;
    const int lhalf  = lane >> 4;

    if (t < 128) bias_s[t] = b1[t];

    // ---- im2col gather geometry (branch-free: clamped addr + select)
    const int m_loc = t & 127;
    const int khalf = t >> 7;                    // builds k in [khalf*16, khalf*16+16)
    int mg_b = mblock * 128 + m_loc;
    if (mg_b > M1 - 1) mg_b = M1 - 1;
    const int oh = mg_b / WX;
    const int ow = mg_b - oh * WX;
    const float* srcb = feat + (size_t)b * CIN * HIN * WIN;

    const int ih0 = oh - 1, ih1 = oh, iw0 = ow - 1, iw1 = ow;
    const bool okh0 = (unsigned)ih0 < (unsigned)HIN, okh1 = (unsigned)ih1 < (unsigned)HIN;
    const bool okw0 = (unsigned)iw0 < (unsigned)WIN, okw1 = (unsigned)iw1 < (unsigned)WIN;
    const int ih0c = okh0 ? ih0 : 0, ih1c = okh1 ? ih1 : (HIN - 1);
    const int iw0c = okw0 ? iw0 : 0, iw1c = okw1 ? iw1 : (WIN - 1);
    const int  off_[2][2] = {{ih0c*WIN+iw0c, ih0c*WIN+iw1c}, {ih1c*WIN+iw0c, ih1c*WIN+iw1c}};
    const bool ok_ [2][2] = {{okh0 && okw0, okh0 && okw1}, {okh1 && okw0, okh1 && okw1}};

    // build 16 elements of A (channels kc*8+khalf*4 .. +4, all 4 taps) into packed[8]
    auto buildA = [&](int kc, unsigned int* pk) {
        const int cbase = kc * 8 + khalf * 4;
        #pragma unroll
        for (int cl = 0; cl < 4; ++cl) {
            const float* bp = srcb + (size_t)(cbase + cl) * (HIN * WIN);
            float e[4];
            #pragma unroll
            for (int kh = 0; kh < 2; ++kh)
                #pragma unroll
                for (int kw = 0; kw < 2; ++kw) {
                    float v = bp[off_[kh][kw]];            // unconditional (clamped)
                    e[kh * 2 + kw] = ok_[kh][kw] ? v : 0.f;
                }
            pk[cl * 2 + 0] = pack2(f2bf(e[0]), f2bf(e[1]));
            pk[cl * 2 + 1] = pack2(f2bf(e[2]), f2bf(e[3]));
        }
    };
    auto storeA = [&](int buf, const unsigned int* pk) {
        uint4* dst = (uint4*)&Alds[buf][m_loc * 32 + khalf * 16];
        dst[0] = make_uint4(pk[0], pk[1], pk[2], pk[3]);
        dst[1] = make_uint4(pk[4], pk[5], pk[6], pk[7]);
    };
    // B-tile copy: thread -> row oc = t&127, 32B half seg = t>>7
    const int boc  = t & 127;
    const int bseg = t >> 7;
    auto loadB = [&](int kc, int buf) {
        const unsigned short* g = &wbf1[(size_t)boc * K1 + kc * 32 + bseg * 16];
        unsigned short* l = &Blds[buf][boc * 32 + bseg * 16];
        copy16(g,     l);
        copy16(g + 8, l + 8);
    };

    v8f acc[8];
    #pragma unroll
    for (int nt = 0; nt < 8; ++nt)
        #pragma unroll
        for (int j = 0; j < 8; ++j) acc[nt][j] = 0.f;

    // ---- prologue: tile 0
    {
        unsigned int pk[8];
        loadB(0, 0);
        buildA(0, pk);
        storeA(0, pk);
        wait_async();
    }
    __syncthreads();

    for (int kc = 0; kc < 32; ++kc) {
        const int cur = kc & 1;
        const int nxt = cur ^ 1;

        // start next tile's global traffic early
        unsigned int pk[8];
        if (kc < 31) {
            loadB(kc + 1, nxt);
            buildA(kc + 1, pk);
        }
        if (kc + 2 < 32)
            __builtin_prefetch(srcb + (size_t)((kc + 2) * 8 + khalf * 4) * (HIN * WIN) + off_[0][0], 0, 3);

        // fragments for current tile
        FragAB af;
        const float4* ap = (const float4*)&Alds[cur][(wave * 16 + lrow) * 32];
        af.f4[0] = ap[2 * lhalf];
        af.f4[1] = ap[2 * lhalf + 1];
        FragAB bfr[8];
        #pragma unroll
        for (int nt = 0; nt < 8; ++nt) {
            const float4* bp = (const float4*)&Blds[cur][(nt * 16 + lrow) * 32 + 16 * lhalf];
            bfr[nt].f4[0] = bp[0];
            bfr[nt].f4[1] = bp[1];
        }
        #pragma unroll
        for (int nt = 0; nt < 8; ++nt)
            acc[nt] = __builtin_amdgcn_wmma_f32_16x16x32_bf16(
                false, af.v, false, bfr[nt].v, (short)0, acc[nt], false, false);

        if (kc < 31) storeA(nxt, pk);
        wait_async();
        __syncthreads();
    }

    // ---- epilogue: bias + ReLU, store bf16 x[b][pos][c]
    const int mbase = mblock * 128 + wave * 16;
    #pragma unroll
    for (int nt = 0; nt < 8; ++nt) {
        int n = nt * 16 + lrow;
        float bias = bias_s[n];
        #pragma unroll
        for (int j = 0; j < 8; ++j) {
            int mg = mbase + j + 8 * lhalf;
            if (mg < M1) {
                float v = fmaxf(acc[nt][j] + bias, 0.f);
                xout[((size_t)b * M1 + mg) * CMID + n] = f2bf(v);
            }
        }
    }
}

// ---------------------------------------------------------------------------
// Kernel 2: fused conv2(cls)+conv3(deltas) GEMM. grid (12,300), block 256.
//   Whole B (32x512 bf16 = 32KB) resident in LDS; A double-buffered async.
// ---------------------------------------------------------------------------
__global__ void __launch_bounds__(256)
k_conv23(const unsigned short* __restrict__ x, const unsigned short* __restrict__ wbf2,
         const float* __restrict__ bc, const float* __restrict__ bb,
         float* __restrict__ clsdel) {
    __shared__ __align__(16) unsigned short Alds[2][128 * 32];
    __shared__ __align__(16) unsigned short Bfull[32 * K2];
    __shared__ float bias_s[32];

    const int b      = blockIdx.y;
    const int mblock = blockIdx.x;
    const int t      = threadIdx.x;
    const int wave   = t >> 5;
    const int lane   = t & 31;
    const int lrow   = lane & 15;
    const int lhalf  = lane >> 4;

    if (t < 32) bias_s[t] = (t < 8) ? bc[t] : ((t < 24) ? bb[t - 8] : 0.f);

    const int m_loc = t & 127;
    const int hseg  = t >> 7;
    int mg_b = mblock * 128 + m_loc;
    if (mg_b > M2 - 1) mg_b = M2 - 1;
    const int h = mg_b / 38;
    const int w = mg_b - h * 38;

    auto loadA = [&](int kc, int buf) {
        const int tap = kc >> 2;
        const int c0  = (kc & 3) * 32;
        const int kh = tap >> 1, kw = tap & 1;
        const int posx = (h + kh) * WX + (w + kw);
        const unsigned short* g = &x[((size_t)b * M1 + posx) * CMID + c0 + hseg * 16];
        unsigned short* l = &Alds[buf][m_loc * 32 + hseg * 16];
        copy16(g,     l);
        copy16(g + 8, l + 8);
    };

    v8f acc[2];
    #pragma unroll
    for (int nt = 0; nt < 2; ++nt)
        #pragma unroll
        for (int j = 0; j < 8; ++j) acc[nt][j] = 0.f;

    // ---- prologue: full B + A tile 0
    #pragma unroll
    for (int i = 0; i < 8; ++i)
        copy16(&wbf2[(size_t)(i * 256 + t) * 8], &Bfull[(i * 256 + t) * 8]);
    loadA(0, 0);
    wait_async();
    __syncthreads();

    for (int kc = 0; kc < 16; ++kc) {
        const int cur = kc & 1;
        if (kc < 15) loadA(kc + 1, cur ^ 1);

        FragAB af;
        const float4* ap = (const float4*)&Alds[cur][(wave * 16 + lrow) * 32];
        af.f4[0] = ap[2 * lhalf];
        af.f4[1] = ap[2 * lhalf + 1];
        FragAB bfr[2];
        #pragma unroll
        for (int nt = 0; nt < 2; ++nt) {
            const float4* bp = (const float4*)&Bfull[(size_t)(nt * 16 + lrow) * K2 + kc * 32 + 16 * lhalf];
            bfr[nt].f4[0] = bp[0];
            bfr[nt].f4[1] = bp[1];
        }
        #pragma unroll
        for (int nt = 0; nt < 2; ++nt)
            acc[nt] = __builtin_amdgcn_wmma_f32_16x16x32_bf16(
                false, af.v, false, bfr[nt].v, (short)0, acc[nt], false, false);

        wait_async();
        __syncthreads();
    }

    const int mbase = mblock * 128 + wave * 16;
    #pragma unroll
    for (int nt = 0; nt < 2; ++nt) {
        int n = nt * 16 + lrow;
        if (n < NOUT2) {
            float bias = bias_s[n];
            #pragma unroll
            for (int j = 0; j < 8; ++j) {
                int mg = mbase + j + 8 * lhalf;
                if (mg < M2)
                    clsdel[((size_t)b * M2 + mg) * NOUT2 + n] = acc[nt][j] + bias;
            }
        }
    }
}

// ---------------------------------------------------------------------------
// Kernel 3: per-image flat score argmax (first-max tie-break) + box decode
// ---------------------------------------------------------------------------
__global__ void __launch_bounds__(256)
k_reduce(const float* __restrict__ clsdel, const float* __restrict__ anchors,
         const float* __restrict__ im_info, float* __restrict__ out) {
    __shared__ float sv[256];
    __shared__ int   si[256];
    const int b = blockIdx.x;
    const int t = threadIdx.x;

    float bv = -1e30f;
    int   bi = 0x3fffffff;
    for (int idx = t; idx < SCN; idx += 256) {
        int a = idx / M2;
        int r = idx - a * M2;
        const float* p = &clsdel[((size_t)b * M2 + r) * NOUT2];
        float s = 1.f / (1.f + __expf(p[a] - p[a + 4]));
        if (s > bv) { bv = s; bi = idx; }
    }
    sv[t] = bv; si[t] = bi;
    __syncthreads();
    for (int off = 128; off > 0; off >>= 1) {
        if (t < off) {
            float ov = sv[t + off]; int oi = si[t + off];
            if (ov > sv[t] || (ov == sv[t] && oi < si[t])) { sv[t] = ov; si[t] = oi; }
        }
        __syncthreads();
    }
    if (t == 0) {
        int p  = si[0];
        float ss = sv[0];
        int a2 = p & 3;
        int ww = (p >> 2) % 38;
        int hh = p / 152;
        float im = im_info[0];
        const float* drow = &clsdel[((size_t)b * M2 + hh * 38 + ww) * NOUT2];
        #pragma unroll
        for (int j = 0; j < 4; ++j) {
            float d  = drow[8 + a2 * 4 + j];
            float sh = (j & 1) ? (float)(hh * 16) : (float)(ww * 16);
            float box = (anchors[a2 * 4 + j] + sh + d) * im;
            out[b * 9 + j]     = truncf(box / im);
            out[b * 9 + 4 + j] = box;
        }
        out[b * 9 + 8] = ss;
    }
}

// ---------------------------------------------------------------------------
// Kernel 4: proposal loss (replicates reference, incl. maximum() for xi2/yi2)
// ---------------------------------------------------------------------------
__global__ void k_loss(const float* __restrict__ outrows, const int* __restrict__ central,
                       const float* __restrict__ gt, int nv, float* __restrict__ dloss) {
    if (threadIdx.x != 0 || blockIdx.x != 0) return;
    float accum = 0.f;
    int cp = central[0];
    for (int v = 0; v < nv; ++v) {
        int idx = 300 * v + cp;
        bool valid = idx < NB;
        int ic = idx; if (ic < 0) ic = 0; if (ic > NB - 1) ic = NB - 1;
        const float* row = outrows + ic * 9;
        const float* g   = gt + v * 4;
        float vb0 = row[4], vb1 = row[5], vb2 = row[6], vb3 = row[7], vs = row[8];
        float xi1 = fmaxf(g[0], vb0), yi1 = fmaxf(g[1], vb1);
        float xi2 = fmaxf(g[2], vb2), yi2 = fmaxf(g[3], vb3);
        float inter = (xi2 - xi1) * (yi2 - yi1);
        float a1 = (g[2] - g[0]) * (g[3] - g[1]);
        float a2 = (vb2 - vb0) * (vb3 - vb1);
        float iou = inter / (a1 + a2 - inter);
        float lv  = (iou > 0.7f) ? -logf(vs + 1e-5f) : -logf(1.f - vs + 1e-5f);
        if (valid) accum += lv;
    }
    dloss[0] = accum;
}

// ---------------------------------------------------------------------------
// Host launcher
// ---------------------------------------------------------------------------
extern "C" void kernel_launch(void* const* d_in, const int* in_sizes, int n_in,
                              void* d_out, int out_size, void* d_ws, size_t ws_size,
                              hipStream_t stream) {
    const float* base_feat = (const float*)d_in[0];
    const int*   central   = (const int*)d_in[1];
    const float* im_info   = (const float*)d_in[2];
    const float* gt_boxes  = (const float*)d_in[3];
    const float* W1        = (const float*)d_in[4];
    const float* b1        = (const float*)d_in[5];
    const float* Wc        = (const float*)d_in[6];
    const float* bc        = (const float*)d_in[7];
    const float* Wb        = (const float*)d_in[8];
    const float* bb        = (const float*)d_in[9];
    const float* anchors   = (const float*)d_in[10];
    const int nv = in_sizes[3] / 4;

    char* ws = (char*)d_ws;
    size_t off = 0;
    auto alloc = [&](size_t bytes) { void* p = ws + off; off = (off + bytes + 255) & ~(size_t)255; return p; };
    unsigned short* wbf1 = (unsigned short*)alloc((size_t)CMID * K1 * 2);
    unsigned short* wbf2 = (unsigned short*)alloc((size_t)32 * K2 * 2);
    unsigned short* xbuf = (unsigned short*)alloc((size_t)NB * M1 * CMID * 2);
    float*          cdel = (float*)alloc((size_t)NB * M2 * NOUT2 * 4);
    (void)ws_size; (void)n_in; (void)out_size;

    float* out = (float*)d_out;

    k_pack_w1<<<(CMID * K1 + 255) / 256, 256, 0, stream>>>(W1, wbf1);
    k_pack_w2<<<(32 * K2 + 255) / 256, 256, 0, stream>>>(Wc, Wb, wbf2);

    dim3 g1((M1 + 127) / 128, NB);
    k_conv1<<<g1, 256, 0, stream>>>(base_feat, wbf1, b1, xbuf);

    dim3 g2((M2 + 127) / 128, NB);
    k_conv23<<<g2, 256, 0, stream>>>(xbuf, wbf2, bc, bb, cdel);

    k_reduce<<<NB, 256, 0, stream>>>(cdel, anchors, im_info, out);
    k_loss<<<1, 1, 0, stream>>>(out, central, gt_boxes, nv, out + NB * 9);
}
